// Finetuner_69707319214472
// MI455X (gfx1250) — compile-verified, matrix operand order checked by reading, not run on hardware
//
#include <hip/hip_runtime.h>
#include <hip/hip_bf16.h>

// ---------------------------------------------------------------------------
// GIN forward (2 layers) for gfx1250 / MI455X.
//   - Node MLPs run as f16 WMMA (v_wmma_f32_16x16x32_f16), weights pre-packed
//     into the CDNA5 B-fragment VGPR layout (CDNA5 ISA §7.12.2) so each
//     lane loads its 16 halves as one contiguous 32B global load.
//   - A-fragments are built from a 16x256 f16 LDS tile with two 16B ds loads.
//   - Edge aggregation / pooling are atomic-add scatter (bandwidth bound).
// ---------------------------------------------------------------------------

typedef __attribute__((ext_vector_type(16))) _Float16 v16h;
typedef __attribute__((ext_vector_type(8)))  _Float16 v8h;
typedef __attribute__((ext_vector_type(8)))  float    v8f;

#define H128 128
#define H2   256
#define NGR  256
#define NTASK 40

// ---- pack a KxNc row-major f32 weight matrix into WMMA B-fragment layout ---
// P[((t*ntiles + nt)*32 + lane)*16 + e] = (f16) W[k*Nc + n]
//   n = nt*16 + (lane&15)
//   k = 32*t + ((e>=8)?16:0) + (lane>>4)*8 + (e&7)      (ISA 16-bit B layout)
__global__ void pack_w_kernel(const float* __restrict__ W, _Float16* __restrict__ P,
                              int K, int Nc) {
    int idx = blockIdx.x * blockDim.x + threadIdx.x;
    if (idx >= K * Nc) return;
    int e    = idx & 15;
    int lane = (idx >> 4) & 31;
    int rest = idx >> 9;
    int ntiles = Nc >> 4;
    int t  = rest / ntiles;
    int nt = rest - t * ntiles;
    int grp = lane >> 4;
    int k = 32 * t + ((e >= 8) ? 16 : 0) + grp * 8 + (e & 7);
    int n = nt * 16 + (lane & 15);
    P[idx] = (_Float16)W[k * Nc + n];
}

// ---- fold BatchNorm (eval) + b1 into per-channel scale a / shift c ----------
__global__ void bnfuse_kernel(const float* __restrict__ g, const float* __restrict__ be,
                              const float* __restrict__ mn, const float* __restrict__ vr,
                              const float* __restrict__ b1,
                              float* __restrict__ a, float* __restrict__ c) {
    int j = blockIdx.x * blockDim.x + threadIdx.x;
    if (j < H2) {
        float s = g[j] * rsqrtf(vr[j] + 1e-5f);
        a[j] = s;
        c[j] = (b1[j] - mn[j]) * s + be[j];
    }
}

// ---- self-loop encoded edge feature: slE = type * enc_w[idx,:] + enc_b -----
__global__ void sloop_kernel(const float* __restrict__ enc_w, const float* __restrict__ enc_b,
                             const int* __restrict__ sli, const int* __restrict__ slt,
                             float* __restrict__ slE) {
    int j = threadIdx.x;
    if (j < H128) slE[j] = (float)(*slt) * enc_w[(*sli) * H128 + j] + enc_b[j];
}

// ---- h0 = node_emb[x] ------------------------------------------------------
__global__ void embed_kernel(const int* __restrict__ x, const float* __restrict__ emb,
                             float* __restrict__ h, int N) {
    long idx = (long)blockIdx.x * blockDim.x + threadIdx.x;
    long i = idx >> 7;
    int  j = (int)(idx & 127);
    if (i < N) h[idx] = emb[x[i] * H128 + j];
}

// ---- agg[i] = [ h[i] , slE ]  (self-loop contribution folded in) -----------
__global__ void init_agg_kernel(const float* __restrict__ h, const float* __restrict__ slE,
                                float* __restrict__ agg, int N) {
    long idx = (long)blockIdx.x * blockDim.x + threadIdx.x;
    long i = idx >> 8;
    int  j = (int)(idx & 255);
    if (i < N) agg[idx] = (j < H128) ? h[i * H128 + j] : slE[j - H128];
}

// ---- edge scatter: agg[dst] += [ h[src], edge_attr @ enc_w + enc_b ] -------
// one wave32 per edge, 8 edges per 256-thread block; enc_w staged in LDS
__global__ __launch_bounds__(256)
void edge_kernel(const int* __restrict__ ei, const float* __restrict__ ea,
                 const float* __restrict__ encw, const float* __restrict__ encb,
                 const float* __restrict__ h, float* __restrict__ agg, int E) {
    __shared__ float wl[9 * H128];
    __shared__ float bl[H128];
    for (int i = threadIdx.x; i < 9 * H128; i += 256) wl[i] = encw[i];
    if (threadIdx.x < H128) bl[threadIdx.x] = encb[threadIdx.x];
    __syncthreads();

    int lane = threadIdx.x & 31;
    int e = blockIdx.x * 8 + (threadIdx.x >> 5);   // wave32: 8 waves / block
    if (e >= E) return;
    int dst = ei[e];
    int src = ei[(long)E + e];
    float att[9];
#pragma unroll
    for (int f = 0; f < 9; ++f) att[f] = ea[(long)e * 9 + f];
    const float* hs = &h[(long)src * H128];
    float* ag = &agg[(long)dst * H2];
#pragma unroll
    for (int k = 0; k < 4; ++k) {
        int j = lane + 32 * k;
        atomicAdd(&ag[j], hs[j]);
        float ev = bl[j];
#pragma unroll
        for (int f = 0; f < 9; ++f) ev += att[f] * wl[f * H128 + j];
        atomicAdd(&ag[H128 + j], ev);
    }
}

// ---- fused node MLP: h = (relu?)( relu(BN(agg@w1)) @ w2 + b2 ) -------------
// 1 block = 16 nodes, 256 threads = 8 wave32s, 24 WMMAs per wave.
__global__ __launch_bounds__(256)
void gemm_bn_kernel(const float* __restrict__ agg,
                    const _Float16* __restrict__ w1p,
                    const float* __restrict__ abn, const float* __restrict__ cbn,
                    const _Float16* __restrict__ w2p, const float* __restrict__ b2,
                    float* __restrict__ hout, int N, int outRelu) {
    __shared__ __align__(16) _Float16 abuf[16 * H2];   // agg tile (f16)
    __shared__ __align__(16) _Float16 zbuf[16 * H2];   // hidden tile (f16)

    int tid  = threadIdx.x;
    int lane = tid & 31;
    int wave = tid >> 5;           // 0..7 (wave32)
    int m15  = lane & 15;
    int grp  = lane >> 4;
    long base = (long)blockIdx.x * 16;

    __builtin_prefetch(w1p, 0, 0);
    __builtin_prefetch(w2p, 0, 0);

    // stage agg tile -> f16 LDS (coalesced: 16-float (64B) chunks per thread)
    {
        int r  = tid >> 4;
        int cs = (tid & 15) * 16;
        long row = base + r;
#pragma unroll
        for (int q = 0; q < 16; ++q) {
            float v = (row < N) ? agg[row * H2 + cs + q] : 0.f;
            abuf[r * H2 + cs + q] = (_Float16)v;
        }
    }
    __syncthreads();

    // ---------------- GEMM1: [16,256] x [256,256], 2 n-tiles / wave ---------
    v8f acc0 = {}; v8f acc1 = {};
    int nt0 = wave * 2, nt1 = wave * 2 + 1;
#pragma unroll
    for (int t = 0; t < 8; ++t) {
        union { v16h v; struct { v8h lo; v8h hi; } s; } A, B0, B1;
        const _Float16* ap = &abuf[m15 * H2 + t * 32 + grp * 8];
        A.s.lo = *(const v8h*)(ap);        // ds_load_b128
        A.s.hi = *(const v8h*)(ap + 16);
        B0.v = *(const v16h*)(w1p + ((size_t)(t * 16 + nt0) * 32 + lane) * 16);
        B1.v = *(const v16h*)(w1p + ((size_t)(t * 16 + nt1) * 32 + lane) * 16);
        acc0 = __builtin_amdgcn_wmma_f32_16x16x32_f16(false, A.v, false, B0.v,
                                                      (short)0, acc0, false, false);
        acc1 = __builtin_amdgcn_wmma_f32_16x16x32_f16(false, A.v, false, B1.v,
                                                      (short)0, acc1, false, false);
    }
    // fused BN (scale/shift) + ReLU -> zbuf (f16)
    {
        int n0 = nt0 * 16 + m15, n1 = nt1 * 16 + m15;
        float a0 = abn[n0], c0 = cbn[n0];
        float a1 = abn[n1], c1 = cbn[n1];
#pragma unroll
        for (int r = 0; r < 8; ++r) {
            int m = r + grp * 8;                       // ISA C/D layout
            zbuf[m * H2 + n0] = (_Float16)fmaxf(acc0[r] * a0 + c0, 0.f);
            zbuf[m * H2 + n1] = (_Float16)fmaxf(acc1[r] * a1 + c1, 0.f);
        }
    }
    __syncthreads();

    // ---------------- GEMM2: [16,256] x [256,128], 1 n-tile / wave ----------
    v8f acc = {};
#pragma unroll
    for (int t = 0; t < 8; ++t) {
        union { v16h v; struct { v8h lo; v8h hi; } s; } A, B;
        const _Float16* ap = &zbuf[m15 * H2 + t * 32 + grp * 8];
        A.s.lo = *(const v8h*)(ap);
        A.s.hi = *(const v8h*)(ap + 16);
        B.v = *(const v16h*)(w2p + ((size_t)(t * 8 + wave) * 32 + lane) * 16);
        acc = __builtin_amdgcn_wmma_f32_16x16x32_f16(false, A.v, false, B.v,
                                                     (short)0, acc, false, false);
    }
    {
        int n = wave * 16 + m15;
        float bb = b2[n];
#pragma unroll
        for (int r = 0; r < 8; ++r) {
            long row = base + r + grp * 8;
            if (row < N) {
                float v = acc[r] + bb;
                if (outRelu) v = fmaxf(v, 0.f);
                hout[row * H128 + n] = v;
            }
        }
    }
}

// ---- pooling + prediction head ---------------------------------------------
__global__ void zero_kernel(float* __restrict__ p, int n) {
    int i = blockIdx.x * blockDim.x + threadIdx.x;
    if (i < n) p[i] = 0.f;
}

__global__ void pool_kernel(const float* __restrict__ h, const int* __restrict__ batch,
                            float* __restrict__ sums, float* __restrict__ cnts, int N) {
    long idx = (long)blockIdx.x * blockDim.x + threadIdx.x;
    long i = idx >> 7;
    int  j = (int)(idx & 127);
    if (i < N) {
        int g = batch[i];
        atomicAdd(&sums[(long)g * H128 + j], h[idx]);
        if (j == 0) atomicAdd(&cnts[g], 1.f);
    }
}

// graph_rep = [pooled, pooled]  =>  out = pooled @ (pw[0:128]+pw[128:256]) + pb
__global__ void pred_kernel(const float* __restrict__ sums, const float* __restrict__ cnts,
                            const float* __restrict__ pw, const float* __restrict__ pb,
                            float* __restrict__ out) {
    int g = blockIdx.x;
    int t = threadIdx.x;
    if (t >= NTASK) return;
    float inv = 1.f / fmaxf(cnts[g], 1.f);
    float acc = pb[t];
#pragma unroll 4
    for (int n = 0; n < H128; ++n) {
        float p = sums[(long)g * H128 + n] * inv;
        acc += p * (pw[n * NTASK + t] + pw[(H128 + n) * NTASK + t]);
    }
    out[g * NTASK + t] = acc;
}

// ---------------------------------------------------------------------------
extern "C" void kernel_launch(void* const* d_in, const int* in_sizes, int n_in,
                              void* d_out, int out_size, void* d_ws, size_t ws_size,
                              hipStream_t stream) {
    const int N = in_sizes[0];
    const int E = in_sizes[1] / 2;

    const int*   x     = (const int*)d_in[0];
    const int*   ei    = (const int*)d_in[1];
    const float* ea    = (const float*)d_in[2];
    const int*   batch = (const int*)d_in[3];
    const int*   sli   = (const int*)d_in[4];
    const int*   slt   = (const int*)d_in[5];
    const float* nemb  = (const float*)d_in[6];

    const float* enc_w[2] = { (const float*)d_in[7],  (const float*)d_in[17] };
    const float* enc_b[2] = { (const float*)d_in[8],  (const float*)d_in[18] };
    const float* w1[2]    = { (const float*)d_in[9],  (const float*)d_in[19] };
    const float* b1[2]    = { (const float*)d_in[10], (const float*)d_in[20] };
    const float* gam[2]   = { (const float*)d_in[11], (const float*)d_in[21] };
    const float* bet[2]   = { (const float*)d_in[12], (const float*)d_in[22] };
    const float* mean[2]  = { (const float*)d_in[13], (const float*)d_in[23] };
    const float* var[2]   = { (const float*)d_in[14], (const float*)d_in[24] };
    const float* w2[2]    = { (const float*)d_in[15], (const float*)d_in[25] };
    const float* b2[2]    = { (const float*)d_in[16], (const float*)d_in[26] };
    const float* pred_w   = (const float*)d_in[27];
    const float* pred_b   = (const float*)d_in[28];
    float* out = (float*)d_out;

    // workspace partition (256B aligned slots)
    char* wsb = (char*)d_ws;
    size_t off = 0;
    auto take = [&](size_t bytes) -> char* {
        char* p = wsb + off;
        off += (bytes + 255) & ~(size_t)255;
        return p;
    };
    float*    h_buf = (float*)take((size_t)N * H128 * 4);
    float*    agg   = (float*)take((size_t)N * H2 * 4);
    _Float16* w1p[2] = { (_Float16*)take(H2 * H2 * 2),  (_Float16*)take(H2 * H2 * 2) };
    _Float16* w2p[2] = { (_Float16*)take(H2 * H128 * 2), (_Float16*)take(H2 * H128 * 2) };
    float* abn[2] = { (float*)take(H2 * 4), (float*)take(H2 * 4) };
    float* cbn[2] = { (float*)take(H2 * 4), (float*)take(H2 * 4) };
    float* slE[2] = { (float*)take(H128 * 4), (float*)take(H128 * 4) };
    float* sums = (float*)take((size_t)NGR * H128 * 4);
    float* cnts = (float*)take(NGR * 4);
    (void)ws_size; (void)n_in; (void)out_size;

    // ---- parameter preprocessing (tiny, once per launch) ----
    for (int l = 0; l < 2; ++l) {
        pack_w_kernel<<<(H2 * H2 + 255) / 256, 256, 0, stream>>>(w1[l], w1p[l], H2, H2);
        pack_w_kernel<<<(H2 * H128 + 255) / 256, 256, 0, stream>>>(w2[l], w2p[l], H2, H128);
        bnfuse_kernel<<<1, H2, 0, stream>>>(gam[l], bet[l], mean[l], var[l], b1[l],
                                            abn[l], cbn[l]);
        sloop_kernel<<<1, H128, 0, stream>>>(enc_w[l], enc_b[l], sli, slt, slE[l]);
    }

    // ---- h0 = node_emb[x] ----
    embed_kernel<<<(int)(((long)N * H128 + 255) / 256), 256, 0, stream>>>(x, nemb, h_buf, N);

    // ---- 2 GIN layers ----
    for (int l = 0; l < 2; ++l) {
        init_agg_kernel<<<(int)(((long)N * H2 + 255) / 256), 256, 0, stream>>>(
            h_buf, slE[l], agg, N);
        edge_kernel<<<(E + 7) / 8, 256, 0, stream>>>(
            ei, ea, enc_w[l], enc_b[l], h_buf, agg, E);
        gemm_bn_kernel<<<(N + 15) / 16, 256, 0, stream>>>(
            agg, w1p[l], abn[l], cbn[l], w2p[l], b2[l], h_buf, N, (l == 0) ? 1 : 0);
    }

    // ---- graph mean pooling + prediction ----
    zero_kernel<<<(NGR * H128 + NGR + 255) / 256, 256, 0, stream>>>(sums, NGR * H128 + NGR);
    // note: cnts immediately follows sums in the same aligned slot? No — separate slot:
    zero_kernel<<<1, 256, 0, stream>>>(cnts, NGR);
    pool_kernel<<<(int)(((long)N * H128 + 255) / 256), 256, 0, stream>>>(
        h_buf, batch, sums, cnts, N);
    pred_kernel<<<NGR, 64, 0, stream>>>(sums, cnts, pred_w, pred_b, out);
}
